// SResTransformerTrain_45810121179212
// MI455X (gfx1250) — matrix operand order repaired and checked by reading, not compiled, not run on hardware
//
#include <hip/hip_runtime.h>
#include <hip/hip_bf16.h>
#include <math.h>

// ---------------- problem constants ----------------
#define BB      2
#define NN      2048
#define DD      256
#define HH      8
#define DQQ     32
#define LL      8
#define DFF     1024
#define MTOK    (BB * NN)          // 4096 tokens
#define CHUNK   32                 // scan chunk length
#define NCH     (NN / CHUNK)       // 64 chunks per sequence
#define EPS_ATTN 1e-6f
#define EPS_LN   1e-5f

typedef __attribute__((ext_vector_type(16))) _Float16 v16h;
typedef __attribute__((ext_vector_type(8)))  float    v8f;
typedef __attribute__((ext_vector_type(4)))  int      v4i;

// activation codes for the GEMM epilogue
#define ACT_NONE 0
#define ACT_RELU 1
#define ACT_ELU1 2   // elu(x)+1  ==  x>0 ? x+1 : exp(x)

// ---------------------------------------------------------------------------
// Optional CDNA5 async global->LDS path (ASYNCcnt-tracked, bypasses VGPRs).
// Guarded so the kernel still compiles on toolchains without the builtins.
// Signature (probe-learned): (int4 AS1*, int4 AS3*, imm int, imm int)
// ---------------------------------------------------------------------------
#if defined(__has_builtin)
#if __has_builtin(__builtin_amdgcn_global_load_async_to_lds_b128) && \
    __has_builtin(__builtin_amdgcn_s_wait_asynccnt)
#define USE_ASYNC_LDS 1
#endif
#endif
#ifndef USE_ASYNC_LDS
#define USE_ASYNC_LDS 0
#endif

#if USE_ASYNC_LDS
typedef __attribute__((address_space(1))) v4i* gv4i_p;
typedef __attribute__((address_space(3))) v4i* lv4i_p;
__device__ __forceinline__ void async_copy16(const _Float16* g, _Float16* l) {
    __builtin_amdgcn_global_load_async_to_lds_b128(
        (gv4i_p)(void*)g, (lv4i_p)(void*)l, 0, 0);
}
#endif

// ---------------------------------------------------------------------------
// Fragment builders from LDS f16 tiles (wave32, 16x16x32 f16 WMMA).
// A (16x32, MxK): lane L holds row M=L&15; lanes 0-15 carry K 0..15,
// lanes 16-31 carry K 16..31 (same k order as B).
// B (32x16, KxN): lane L holds col N=L&15; lanes 0-15 carry K 0..15,
// lanes 16-31 carry K 16..31.
// ---------------------------------------------------------------------------
__device__ __forceinline__ v16h frag_a_from_lds(const _Float16* p, int lane, int ld) {
    const _Float16* row = p + (size_t)(lane & 15) * ld;
    int kb = (lane < 16) ? 0 : 16;
    v16h f;
#pragma unroll
    for (int j = 0; j < 16; ++j) f[j] = row[kb + j];
    return f;
}

__device__ __forceinline__ v16h frag_b_from_lds(const _Float16* p, int lane, int ld) {
    const _Float16* row = p + (size_t)(lane & 15) * ld;
    int kb = (lane < 16) ? 0 : 16;
    v16h f;
#pragma unroll
    for (int j = 0; j < 16; ++j) f[j] = row[kb + j];
    return f;
}

// ---------------------------------------------------------------------------
// Weight convert + transpose:  W[K][Nc] fp32  ->  Wt[Nc][K] f16
// ---------------------------------------------------------------------------
__global__ __launch_bounds__(256) void convert_w_kernel(
    const float* __restrict__ W, _Float16* __restrict__ Wt, int K, int Nc)
{
    __shared__ float tile[32][33];
    const int k0 = blockIdx.x * 32, n0 = blockIdx.y * 32;
    const int tx = threadIdx.x & 31, ty = threadIdx.x >> 5;   // ty 0..7
#pragma unroll
    for (int i = 0; i < 4; ++i) {
        const int k = ty + i * 8;
        tile[k][tx] = W[(size_t)(k0 + k) * Nc + n0 + tx];
    }
    __syncthreads();
#pragma unroll
    for (int i = 0; i < 4; ++i) {
        const int n = ty + i * 8;
        Wt[(size_t)(n0 + n) * K + k0 + tx] = (_Float16)tile[tx][n];
    }
}

// ---------------------------------------------------------------------------
// WMMA GEMM:  C[M,Nc] = epilogue( A[M,K] @ Wt[Nc,K]^T + bias (+ resid) )
// A, Wt already f16 in global. block = 128 threads (4 waves), 64x64 tile,
// K tiled by 64 (two 16x16x32 sub-steps -> 8 v_wmma per wave per stage).
// ---------------------------------------------------------------------------
__global__ __launch_bounds__(128) void gemm_wmma_kernel(
    const _Float16* __restrict__ A, const _Float16* __restrict__ Wt,
    const float* __restrict__ bias, const float* __restrict__ resid,
    float* __restrict__ Cf, _Float16* __restrict__ Ch,
    int M, int Kdim, int Nc, int act)
{
    __shared__ _Float16 As[64][72];   // 64 rows x 64 K (+8 pad), 144B row stride
    __shared__ _Float16 Bs[64][72];

    const int tid  = threadIdx.x;
    const int lane = tid & 31;
    const int w    = tid >> 5;
    const int m0   = blockIdx.x * 64;
    const int n0   = blockIdx.y * 64;

    v8f acc[4];
#pragma unroll
    for (int t = 0; t < 4; ++t)
#pragma unroll
        for (int r = 0; r < 8; ++r) acc[t][r] = 0.0f;

    for (int k0 = 0; k0 < Kdim; k0 += 64) {
        __syncthreads();
        // stage 64x64 f16 tiles: 512 x 16B chunks per tile, 4 per thread
#pragma unroll
        for (int i = 0; i < 4; ++i) {
            const int c = tid + 128 * i;
            const int row = c >> 3;
            const int seg = c & 7;
#if USE_ASYNC_LDS
            async_copy16(A  + (size_t)(m0 + row) * Kdim + k0 + seg * 8, &As[row][seg * 8]);
            async_copy16(Wt + (size_t)(n0 + row) * Kdim + k0 + seg * 8, &Bs[row][seg * 8]);
#else
            *(uint4*)(&As[row][seg * 8]) =
                *(const uint4*)(A  + (size_t)(m0 + row) * Kdim + k0 + seg * 8);
            *(uint4*)(&Bs[row][seg * 8]) =
                *(const uint4*)(Wt + (size_t)(n0 + row) * Kdim + k0 + seg * 8);
#endif
        }
#if USE_ASYNC_LDS
        __builtin_amdgcn_s_wait_asynccnt(0);
#endif
        __syncthreads();

#pragma unroll
        for (int s = 0; s < 2; ++s) {
            const v16h af = frag_a_from_lds(&As[w * 16][s * 32], lane, 72);
#pragma unroll
            for (int t = 0; t < 4; ++t) {
                const v16h bf = frag_b_from_lds(&Bs[t * 16][s * 32], lane, 72);
                acc[t] = __builtin_amdgcn_wmma_f32_16x16x32_f16(
                    false, af, false, bf, (short)0, acc[t], false, false);
            }
        }
    }

    // epilogue: C/D layout -> vgpr r: M = w*16 + r + (lane>=16 ? 8 : 0), N = lane&15
    const int mb = w * 16 + ((lane >> 4) << 3);
#pragma unroll
    for (int t = 0; t < 4; ++t) {
        const int gn = n0 + (lane & 15) + t * 16;
        const float bv = bias ? bias[gn] : 0.0f;
#pragma unroll
        for (int r = 0; r < 8; ++r) {
            const int gm = m0 + mb + r;
            float v = acc[t][r] + bv;
            if (act == ACT_RELU)      v = fmaxf(v, 0.0f);
            else if (act == ACT_ELU1) v = (v > 0.0f) ? (v + 1.0f) : __expf(v);
            if (resid) v += resid[(size_t)gm * Nc + gn];
            if (Cf) Cf[(size_t)gm * Nc + gn] = v;
            if (Ch) Ch[(size_t)gm * Nc + gn] = (_Float16)v;
        }
    }
}

// ---------------------------------------------------------------------------
// Embedding: h[:, :128] = x @ W_emb + b_emb ; h[:, 128:] = pe  (K=2 GEMV)
// Writes fp32 h (for residual/LN) and f16 hh (GEMM operand).
// ---------------------------------------------------------------------------
__global__ __launch_bounds__(128) void embed_kernel(
    const float* __restrict__ x, const float* __restrict__ pe,
    const float* __restrict__ Wemb, const float* __restrict__ bemb,
    float* __restrict__ h, _Float16* __restrict__ hh)
{
    const int t = blockIdx.x;          // token 0..4095
    const int j = threadIdx.x;         // 0..127
    const int n = t & (NN - 1);
    const float x0 = x[(size_t)t * 2 + 0];
    const float x1 = x[(size_t)t * 2 + 1];
    const float e0 = x0 * Wemb[j] + x1 * Wemb[128 + j] + bemb[j];
    const float e1 = pe[(size_t)n * 128 + j];
    h [(size_t)t * DD + j]        = e0;
    h [(size_t)t * DD + 128 + j]  = e1;
    hh[(size_t)t * DD + j]        = (_Float16)e0;
    hh[(size_t)t * DD + 128 + j]  = (_Float16)e1;
}

// ---------------------------------------------------------------------------
// Causal linear attention scan, 3 passes. State per (b,h): S[32][32], Ksum[32].
// Thread m of a wave owns column m of the state; K[d]/Q[d] broadcast via shfl.
// ---------------------------------------------------------------------------
__global__ __launch_bounds__(32) void scan_pass1(
    const float* __restrict__ Kf, const float* __restrict__ Vf,
    float* __restrict__ Ssum, float* __restrict__ Ksum)
{
    const int c  = blockIdx.x;         // chunk
    const int bh = blockIdx.y;         // b*H + h
    const int b  = bh >> 3, hh = bh & 7;
    const int m  = threadIdx.x;

    float S[32];
#pragma unroll
    for (int d = 0; d < 32; ++d) S[d] = 0.0f;
    float ks = 0.0f;

    for (int n = c * CHUNK; n < c * CHUNK + CHUNK; ++n) {
        const size_t idx = (size_t)(b * NN + n) * DD + hh * DQQ + m;
        const float kv = Kf[idx];
        const float vv = Vf[idx];
        ks += kv;
#pragma unroll
        for (int d = 0; d < 32; ++d) S[d] += __shfl(kv, d) * vv;
    }
    const size_t base = ((size_t)bh * NCH + c) * (DQQ * DQQ);
#pragma unroll
    for (int d = 0; d < 32; ++d) Ssum[base + (size_t)d * 32 + m] = S[d];
    Ksum[((size_t)bh * NCH + c) * 32 + m] = ks;
}

__global__ __launch_bounds__(32) void scan_pass2(
    float* __restrict__ Ssum, float* __restrict__ Ksum)
{
    const int bh = blockIdx.x;
    const int m  = threadIdx.x;
    float run[32];
#pragma unroll
    for (int d = 0; d < 32; ++d) run[d] = 0.0f;
    float krun = 0.0f;

    for (int c = 0; c < NCH; ++c) {
        const size_t base = ((size_t)bh * NCH + c) * (DQQ * DQQ) + m;
#pragma unroll
        for (int d = 0; d < 32; ++d) {
            const float t = Ssum[base + (size_t)d * 32];
            Ssum[base + (size_t)d * 32] = run[d];    // exclusive prefix
            run[d] += t;
        }
        const size_t kb = ((size_t)bh * NCH + c) * 32 + m;
        const float tk = Ksum[kb];
        Ksum[kb] = krun;
        krun += tk;
    }
}

__global__ __launch_bounds__(32) void scan_pass3(
    const float* __restrict__ Qf, const float* __restrict__ Kf,
    const float* __restrict__ Vf, const float* __restrict__ Ssum,
    const float* __restrict__ Ksum, _Float16* __restrict__ O)
{
    const int c  = blockIdx.x;
    const int bh = blockIdx.y;
    const int b  = bh >> 3, hh = bh & 7;
    const int m  = threadIdx.x;

    float S[32];
    const size_t base = ((size_t)bh * NCH + c) * (DQQ * DQQ);
#pragma unroll
    for (int d = 0; d < 32; ++d) S[d] = Ssum[base + (size_t)d * 32 + m];
    float kc = Ksum[((size_t)bh * NCH + c) * 32 + m];

    for (int n = c * CHUNK; n < c * CHUNK + CHUNK; ++n) {
        const size_t idx = (size_t)(b * NN + n) * DD + hh * DQQ + m;
        const float qv = Qf[idx];
        const float kv = Kf[idx];
        const float vv = Vf[idx];
        kc += kv;                                   // inclusive cumsum(K)
        float zs = qv * kc;                         // Q . cumsum(K)
#pragma unroll
        for (int off = 16; off > 0; off >>= 1) zs += __shfl_xor(zs, off);
        const float z = 1.0f / (zs + EPS_ATTN);
        float accv = 0.0f;
#pragma unroll
        for (int d = 0; d < 32; ++d) {
            S[d]  += __shfl(kv, d) * vv;            // inclusive cumsum(K (x) V)
            accv  += __shfl(qv, d) * S[d];          // Q . S
        }
        O[idx] = (_Float16)(accv * z);              // feeds out-proj GEMM (f16)
    }
}

// ---------------------------------------------------------------------------
// LayerNorm over d=256 (one token per 256-thread block); dual fp32/f16 output
// ---------------------------------------------------------------------------
__global__ __launch_bounds__(256) void ln_kernel(
    const float* __restrict__ in, const float* __restrict__ g,
    const float* __restrict__ b, float* __restrict__ out,
    _Float16* __restrict__ outh)
{
    __shared__ float s1[256], s2[256];
    const int t = blockIdx.x, j = threadIdx.x;
    const float v = in[(size_t)t * DD + j];
    s1[j] = v; s2[j] = v * v;
    __syncthreads();
#pragma unroll
    for (int off = 128; off > 0; off >>= 1) {
        if (j < off) { s1[j] += s1[j + off]; s2[j] += s2[j + off]; }
        __syncthreads();
    }
    const float mean = s1[0] * (1.0f / DD);
    const float var  = s2[0] * (1.0f / DD) - mean * mean;
    const float r    = rsqrtf(var + EPS_LN);
    const float o    = (v - mean) * r * g[j] + b[j];
    out[(size_t)t * DD + j] = o;
    if (outh) outh[(size_t)t * DD + j] = (_Float16)o;
}

// ---------------------------------------------------------------------------
// Output heads: out[t,0] = h.Wa + ba ; out[t,1] = tanh(h.Wp + bp)
// ---------------------------------------------------------------------------
__global__ __launch_bounds__(256) void head_kernel(
    const float* __restrict__ hn, const float* __restrict__ Wa,
    const float* __restrict__ ba, const float* __restrict__ Wp,
    const float* __restrict__ bp, float* __restrict__ out)
{
    __shared__ float s1[256], s2[256];
    const int t = blockIdx.x, j = threadIdx.x;
    const float v = hn[(size_t)t * DD + j];
    s1[j] = v * Wa[j]; s2[j] = v * Wp[j];
    __syncthreads();
#pragma unroll
    for (int off = 128; off > 0; off >>= 1) {
        if (j < off) { s1[j] += s1[j + off]; s2[j] += s2[j + off]; }
        __syncthreads();
    }
    if (j == 0) {
        out[(size_t)t * 2 + 0] = s1[0] + ba[0];
        out[(size_t)t * 2 + 1] = tanhf(s2[0] + bp[0]);
    }
}

// ---------------------------------------------------------------------------
// Host-side launch sequence
// ---------------------------------------------------------------------------
extern "C" void kernel_launch(void* const* d_in, const int* in_sizes, int n_in,
                              void* d_out, int out_size, void* d_ws, size_t ws_size,
                              hipStream_t stream) {
    const float* x    = (const float*)d_in[0];
    const float* pe   = (const float*)d_in[1];
    const float* Wemb = (const float*)d_in[2];
    const float* bemb = (const float*)d_in[3];
    const float* Wq   = (const float*)d_in[4];
    const float* bq   = (const float*)d_in[5];
    const float* Wk   = (const float*)d_in[6];
    const float* bk   = (const float*)d_in[7];
    const float* Wv   = (const float*)d_in[8];
    const float* bv   = (const float*)d_in[9];
    const float* Wo   = (const float*)d_in[10];
    const float* bo   = (const float*)d_in[11];
    const float* g1   = (const float*)d_in[12];
    const float* be1  = (const float*)d_in[13];
    const float* g2   = (const float*)d_in[14];
    const float* be2  = (const float*)d_in[15];
    const float* W1   = (const float*)d_in[16];
    const float* bf1  = (const float*)d_in[17];
    const float* W2   = (const float*)d_in[18];
    const float* bf2  = (const float*)d_in[19];
    const float* gf   = (const float*)d_in[20];
    const float* bff  = (const float*)d_in[21];
    const float* Wa   = (const float*)d_in[22];
    const float* ba   = (const float*)d_in[23];
    const float* Wp   = (const float*)d_in[24];
    const float* bp   = (const float*)d_in[25];

    // ---------------- workspace layout ----------------
    const size_t tok256 = (size_t)MTOK * DD;        // 1,048,576
    float* p = (float*)d_ws;
    float* h    = p;  p += tok256;                               // fp32 activations
    float* t2   = p;  p += tok256;                               // GEMM+resid out
    float* qb   = p;  p += tok256;
    float* kb   = p;  p += tok256;
    float* vb   = p;  p += tok256;
    float* Ssum = p;  p += (size_t)BB * HH * NCH * DQQ * DQQ;    // 1,048,576
    float* Ksum = p;  p += (size_t)BB * HH * NCH * DQQ;          // 32,768
    _Float16* hp = (_Float16*)p;
    _Float16* hh  = hp;  hp += tok256;                           // f16 of h
    _Float16* abh = hp;  hp += tok256;                           // f16 attn out
    _Float16* t1h = hp;  hp += (size_t)MTOK * DFF;               // f16 FFN mid
    _Float16* Wh  = hp;  hp += (size_t)LL * (4 * DD * DD + 2 * DD * DFF);

    // per-layer f16 weight offsets within Wh
    const size_t LWSZ = 4 * (size_t)DD * DD + 2 * (size_t)DD * DFF;  // 786432
    const size_t OQ = 0, OK = 65536, OV = 131072, OO = 196608, O1 = 262144, O2 = 524288;

    // ---------------- weight convert+transpose (f16, [n][k]) ----------------
    for (int i = 0; i < LL; ++i) {
        _Float16* base = Wh + (size_t)i * LWSZ;
        const size_t wdd = (size_t)i * DD * DD;
        const size_t wdf = (size_t)i * DD * DFF;
        convert_w_kernel<<<dim3(DD / 32, DD / 32), 256, 0, stream>>>(Wq + wdd, base + OQ, DD, DD);
        convert_w_kernel<<<dim3(DD / 32, DD / 32), 256, 0, stream>>>(Wk + wdd, base + OK, DD, DD);
        convert_w_kernel<<<dim3(DD / 32, DD / 32), 256, 0, stream>>>(Wv + wdd, base + OV, DD, DD);
        convert_w_kernel<<<dim3(DD / 32, DD / 32), 256, 0, stream>>>(Wo + wdd, base + OO, DD, DD);
        convert_w_kernel<<<dim3(DD / 32, DFF / 32), 256, 0, stream>>>(W1 + wdf, base + O1, DD, DFF);
        convert_w_kernel<<<dim3(DFF / 32, DD / 32), 256, 0, stream>>>(W2 + wdf, base + O2, DFF, DD);
    }

    const dim3 blkGemm(128);
    const dim3 grdD(MTOK / 64, DD / 64);            // 64 x 4
    const dim3 grdF(MTOK / 64, DFF / 64);           // 64 x 16
    const dim3 grdScan(NCH, BB * HH);               // 64 x 16

    embed_kernel<<<MTOK, 128, 0, stream>>>(x, pe, Wemb, bemb, h, hh);

    for (int i = 0; i < LL; ++i) {
        _Float16* base = Wh + (size_t)i * LWSZ;
        const size_t bd = (size_t)i * DD;
        const size_t bf = (size_t)i * DFF;

        // Q,K with fused elu+1 feature map; V plain (fp32 outputs for the scan)
        gemm_wmma_kernel<<<grdD, blkGemm, 0, stream>>>(hh, base + OQ, bq + bd, nullptr, qb, nullptr, MTOK, DD, DD, ACT_ELU1);
        gemm_wmma_kernel<<<grdD, blkGemm, 0, stream>>>(hh, base + OK, bk + bd, nullptr, kb, nullptr, MTOK, DD, DD, ACT_ELU1);
        gemm_wmma_kernel<<<grdD, blkGemm, 0, stream>>>(hh, base + OV, bv + bd, nullptr, vb, nullptr, MTOK, DD, DD, ACT_NONE);

        // chunked causal prefix scan -> f16 attn output
        scan_pass1<<<grdScan, 32, 0, stream>>>(kb, vb, Ssum, Ksum);
        scan_pass2<<<BB * HH, 32, 0, stream>>>(Ssum, Ksum);
        scan_pass3<<<grdScan, 32, 0, stream>>>(qb, kb, vb, Ssum, Ksum, abh);

        // out projection + residual, then LN1 (-> h fp32 + hh f16)
        gemm_wmma_kernel<<<grdD, blkGemm, 0, stream>>>(abh, base + OO, bo + bd, h, t2, nullptr, MTOK, DD, DD, ACT_NONE);
        ln_kernel<<<MTOK, 256, 0, stream>>>(t2, g1 + bd, be1 + bd, h, hh);

        // FFN: relu GEMM (f16-only out) -> GEMM + residual, then LN2
        gemm_wmma_kernel<<<grdF, blkGemm, 0, stream>>>(hh, base + O1, bf1 + bf, nullptr, nullptr, t1h, MTOK, DD, DFF, ACT_RELU);
        gemm_wmma_kernel<<<grdD, blkGemm, 0, stream>>>(t1h, base + O2, bf2 + bd, h, t2, nullptr, MTOK, DFF, DD, ACT_NONE);
        ln_kernel<<<MTOK, 256, 0, stream>>>(t2, g2 + bd, be2 + bd, h, hh);
    }

    // final LN + heads
    ln_kernel<<<MTOK, 256, 0, stream>>>(h, gf, bff, t2, nullptr);
    head_kernel<<<MTOK, 256, 0, stream>>>(t2, Wa, ba, Wp, bp, (float*)d_out);
}
